// ASAPLipsNet_83116207112614
// MI455X (gfx1250) — matrix operand order repaired and checked by reading, not compiled, not run on hardware
//
#include <hip/hip_runtime.h>
#include <hip/hip_bf16.h>
#include <math.h>

typedef _Float16 f16;
typedef _Float16 h8   __attribute__((ext_vector_type(8)));
typedef _Float16 v16h __attribute__((ext_vector_type(16)));
typedef float    v8f  __attribute__((ext_vector_type(8)));

#define B_D   4096
#define IN_D  128
#define H_D   256
#define OUT_D 64
#define EPSV  1e-4f

// ---------------------------------------------------------------------------
// A-fragment (16x32 f16, M x K) per ISA 7.12.2:
//   lanes 0-15 hold row M=lane, K = 0..7 (VGPR0-3) and 16..23 (VGPR4-7)
//   lanes 16-31 hold row M=lane-16, K = 8..15 and 24..31
// base = row pointer already offset by ks; kb = (lane>>4)*8
// ---------------------------------------------------------------------------
__device__ __forceinline__ v16h make_afrag(h8 lo, h8 hi) {
  v16h a;
#pragma unroll
  for (int i = 0; i < 8; ++i) { a[i] = lo[i]; a[8 + i] = hi[i]; }
  return a;
}

// ---------------------------------------------------------------------------
// Forward layer: out = relu(A @ W^T + bias), store activation + mask (f16).
// A: [M x K] f16 row-major. W: [N x K] f16 row-major (acts as Bt: column n of
// the K x N right matrix is row n of W, contiguous in K).
// One 16x16 output tile per wave; 8 waves per block.
// ---------------------------------------------------------------------------
__global__ __launch_bounds__(256)
void layer_kernel(const f16* __restrict__ A, const f16* __restrict__ W,
                  const float* __restrict__ bias,
                  f16* __restrict__ outA, f16* __restrict__ outM,
                  int M, int N, int K) {
  const int lane = threadIdx.x & 31;
  const int wid  = threadIdx.x >> 5;
  const int nct  = N >> 4;
  const int tile = blockIdx.x * 8 + wid;
  if (tile >= (M >> 4) * nct) return;
  const int bt = tile / nct, ct = tile % nct;

  const int n15  = lane & 15;
  const int kb   = (lane >> 4) << 3;   // A-frag K sub-offset
  const int kb16 = (lane >> 4) << 4;   // unused; B-frag uses kb as well

  const f16* arow = A + (size_t)(bt * 16 + n15) * K;
  const f16* brow = W + (size_t)(ct * 16 + n15) * K;

  v8f acc = {};
  for (int ks = 0; ks < K; ks += 32) {
    h8 alo = *reinterpret_cast<const h8*>(arow + ks + kb);
    h8 ahi = *reinterpret_cast<const h8*>(arow + ks + kb + 16);
    v16h a = make_afrag(alo, ahi);
    v16h b = make_afrag(*reinterpret_cast<const h8*>(brow + ks + kb),
                        *reinterpret_cast<const h8*>(brow + ks + kb + 16));
    acc = __builtin_amdgcn_wmma_f32_16x16x32_f16(false, a, false, b,
                                                 (short)0, acc, false, false);
  }
  const int col  = ct * 16 + n15;
  const float bv = bias[col];
  const int row0 = bt * 16 + ((lane >> 4) << 3);
#pragma unroll
  for (int r = 0; r < 8; ++r) {
    float h = acc[r] + bv;
    float m = h > 0.f ? 1.f : 0.f;
    size_t idx = (size_t)(row0 + r) * N + col;
    outA[idx] = (f16)(h * m);
    outM[idx] = (f16)m;
  }
}

// ---------------------------------------------------------------------------
// Head: fout = A @ Wa^T + ba   (f32 output, no activation)
// ---------------------------------------------------------------------------
__global__ __launch_bounds__(256)
void head_kernel(const f16* __restrict__ A, const f16* __restrict__ W,
                 const float* __restrict__ bias, float* __restrict__ out,
                 int M, int N, int K) {
  const int lane = threadIdx.x & 31;
  const int wid  = threadIdx.x >> 5;
  const int nct  = N >> 4;
  const int tile = blockIdx.x * 8 + wid;
  if (tile >= (M >> 4) * nct) return;
  const int bt = tile / nct, ct = tile % nct;

  const int n15  = lane & 15;
  const int kb   = (lane >> 4) << 3;
  const int kb16 = (lane >> 4) << 4;

  const f16* arow = A + (size_t)(bt * 16 + n15) * K;
  const f16* brow = W + (size_t)(ct * 16 + n15) * K;

  v8f acc = {};
  for (int ks = 0; ks < K; ks += 32) {
    h8 alo = *reinterpret_cast<const h8*>(arow + ks + kb);
    h8 ahi = *reinterpret_cast<const h8*>(arow + ks + kb + 16);
    v16h a = make_afrag(alo, ahi);
    v16h b = make_afrag(*reinterpret_cast<const h8*>(brow + ks + kb),
                        *reinterpret_cast<const h8*>(brow + ks + kb + 16));
    acc = __builtin_amdgcn_wmma_f32_16x16x32_f16(false, a, false, b,
                                                 (short)0, acc, false, false);
  }
  const int col  = ct * 16 + n15;
  const float bv = bias[col];
  const int row0 = bt * 16 + ((lane >> 4) << 3);
#pragma unroll
  for (int r = 0; r < 8; ++r)
    out[(size_t)(row0 + r) * N + col] = acc[r] + bv;
}

// ---------------------------------------------------------------------------
// Per-sample Jacobian Frobenius norm^2.
//   Step A: A_b[o,j] = m2[o] * sum_i W2[o,i]*m1[i]*W1t[j,i]   -> LDS (transposed)
//   Step B: nrm += sum_{p,j} ( m3[p] * sum_o W3[p,o]*A_b[o,j] )^2
// One block (8 waves) per sample; each wave owns 2 of 16 row-tiles.
// ---------------------------------------------------------------------------
#define ATS 264  // padded LDS row stride (halves): 256 + 8, keeps 16B alignment

__global__ __launch_bounds__(256)
void jac_kernel(const f16* __restrict__ W2h, const f16* __restrict__ W3h,
                const f16* __restrict__ W1t, const f16* __restrict__ m1g,
                const f16* __restrict__ m2g, const f16* __restrict__ m3g,
                float* __restrict__ norm2) {
  __shared__ f16   At[128 * ATS];     // A_b transposed: At[j][o]
  __shared__ f16   m1s[H_D];
  __shared__ float m2s[H_D];
  __shared__ float m3s[H_D];
  __shared__ float red[256];

  const int b    = blockIdx.x;
  const int tid  = threadIdx.x;
  const int lane = tid & 31;
  const int wid  = tid >> 5;

  m1s[tid] = m1g[(size_t)b * H_D + tid];
  m2s[tid] = (float)m2g[(size_t)b * H_D + tid];
  m3s[tid] = (float)m3g[(size_t)b * H_D + tid];
  __syncthreads();

  const int n15    = lane & 15;
  const int kb     = (lane >> 4) << 3;
  const int kb16   = (lane >> 4) << 4;
  const int rowsel = (lane >> 4) << 3;

  // ---- Step A: masked W2 @ W1^T, write transposed f16 tile into LDS ----
#pragma unroll
  for (int rr = 0; rr < 2; ++rr) {
    const int rt = wid * 2 + rr;
    const f16* w2row = W2h + (size_t)(rt * 16 + n15) * H_D;
    v8f acc[8];
#pragma unroll
    for (int c = 0; c < 8; ++c) acc[c] = v8f{};

    for (int kk = 0; kk < 8; ++kk) {
      const int ks = kk * 32;
      h8 wlo = *reinterpret_cast<const h8*>(w2row + ks + kb);
      h8 whi = *reinterpret_cast<const h8*>(w2row + ks + kb + 16);
      h8 mlo = *reinterpret_cast<const h8*>(&m1s[ks + kb]);
      h8 mhi = *reinterpret_cast<const h8*>(&m1s[ks + kb + 16]);
      wlo *= mlo;
      whi *= mhi;
      v16h a = make_afrag(wlo, whi);
#pragma unroll
      for (int c = 0; c < 8; ++c) {
        const f16* brow = W1t + (size_t)(c * 16 + n15) * H_D + ks + kb;
        v16h bf = make_afrag(*reinterpret_cast<const h8*>(brow),
                             *reinterpret_cast<const h8*>(brow + 16));
        acc[c] = __builtin_amdgcn_wmma_f32_16x16x32_f16(false, a, false, bf,
                                                        (short)0, acc[c],
                                                        false, false);
      }
    }
    const int row0 = rt * 16 + rowsel;
    float rs[8];
#pragma unroll
    for (int r = 0; r < 8; ++r) rs[r] = m2s[row0 + r];
#pragma unroll
    for (int c = 0; c < 8; ++c) {
      h8 p;
#pragma unroll
      for (int r = 0; r < 8; ++r) p[r] = (f16)(acc[c][r] * rs[r]);
      // transposed store: At[j = c*16+n15][o = row0 .. row0+7], contiguous
      *reinterpret_cast<h8*>(&At[(c * 16 + n15) * ATS + row0]) = p;
    }
  }
  __syncthreads();

  // ---- Step B: W3 @ A_b, mask rows by m3, accumulate sum of squares ----
  float nacc = 0.f;
#pragma unroll
  for (int rr = 0; rr < 2; ++rr) {
    const int rt = wid * 2 + rr;
    const f16* w3row = W3h + (size_t)(rt * 16 + n15) * H_D;
    v8f acc[8];
#pragma unroll
    for (int c = 0; c < 8; ++c) acc[c] = v8f{};

    for (int kk = 0; kk < 8; ++kk) {
      const int ks = kk * 32;
      h8 wlo = *reinterpret_cast<const h8*>(w3row + ks + kb);
      h8 whi = *reinterpret_cast<const h8*>(w3row + ks + kb + 16);
      v16h a = make_afrag(wlo, whi);
#pragma unroll
      for (int c = 0; c < 8; ++c) {
        const f16* lp = &At[(c * 16 + n15) * ATS + ks + kb];
        h8 blo = *reinterpret_cast<const h8*>(lp);
        h8 bhi = *reinterpret_cast<const h8*>(lp + 16);
        v16h bf = make_afrag(blo, bhi);  // table order: k = kb..kb+7, kb+16..kb+23
        acc[c] = __builtin_amdgcn_wmma_f32_16x16x32_f16(false, a, false, bf,
                                                        (short)0, acc[c],
                                                        false, false);
      }
    }
    const int row0 = rt * 16 + rowsel;
    float rs[8];
#pragma unroll
    for (int r = 0; r < 8; ++r) rs[r] = m3s[row0 + r];
#pragma unroll
    for (int c = 0; c < 8; ++c)
#pragma unroll
      for (int r = 0; r < 8; ++r) {
        float v = acc[c][r] * rs[r];
        nacc += v * v;
      }
  }

  red[tid] = nacc;
  for (int s = 128; s > 0; s >>= 1) {
    __syncthreads();
    if (tid < s) red[tid] += red[tid + s];
  }
  if (tid == 0) norm2[b] = red[0];
}

// ---------------------------------------------------------------------------
// f32 -> f16 conversions (+ W1 transpose) and softplus(k)
// ---------------------------------------------------------------------------
__global__ void convert_kernel(const float* __restrict__ x,
                               const float* __restrict__ W1,
                               const float* __restrict__ W2,
                               const float* __restrict__ W3,
                               const float* __restrict__ Wa,
                               const float* __restrict__ k,
                               f16* xh, f16* w1h, f16* w1t, f16* w2h,
                               f16* w3h, f16* wah, float* kval) {
  const int t = blockIdx.x * blockDim.x + threadIdx.x;
  if (t < B_D * IN_D) xh[t] = (f16)x[t];
  if (t < H_D * H_D) { w2h[t] = (f16)W2[t]; w3h[t] = (f16)W3[t]; }
  if (t < H_D * IN_D) {
    float v = W1[t];
    w1h[t] = (f16)v;
    int i = t / IN_D, j = t % IN_D;   // W1[i,j]
    w1t[j * H_D + i] = (f16)v;        // W1t[j,i]
  }
  if (t < OUT_D * H_D) wah[t] = (f16)Wa[t];
  if (t == 0) {
    float kk = k[0];
    kval[0] = (kk > 20.f) ? kk : log1pf(expf(kk));  // softplus
  }
}

// ---------------------------------------------------------------------------
// out = tanh( softplus(k) * fout / (sqrt(norm2) + eps) )
// ---------------------------------------------------------------------------
__global__ void final_kernel(const float* __restrict__ fout,
                             const float* __restrict__ norm2,
                             const float* __restrict__ kval,
                             float* __restrict__ out) {
  const int t = blockIdx.x * blockDim.x + threadIdx.x;
  const int b = t >> 6;  // OUT_D = 64
  float jn = sqrtf(norm2[b]);
  out[t] = tanhf(kval[0] * fout[t] / (jn + EPSV));
}

// ---------------------------------------------------------------------------
extern "C" void kernel_launch(void* const* d_in, const int* in_sizes, int n_in,
                              void* d_out, int out_size, void* d_ws,
                              size_t ws_size, hipStream_t stream) {
  const float* x  = (const float*)d_in[0];
  const float* W1 = (const float*)d_in[1];
  const float* b1 = (const float*)d_in[2];
  const float* W2 = (const float*)d_in[3];
  const float* b2 = (const float*)d_in[4];
  const float* W3 = (const float*)d_in[5];
  const float* b3 = (const float*)d_in[6];
  const float* Wa = (const float*)d_in[7];
  const float* ba = (const float*)d_in[8];
  const float* k  = (const float*)d_in[9];

  char* p = (char*)d_ws;
  auto alloc = [&](size_t bytes) -> char* {
    char* r = p;
    p += (bytes + 255) & ~(size_t)255;
    return r;
  };
  f16* xh    = (f16*)alloc((size_t)B_D * IN_D * 2);
  f16* w1h   = (f16*)alloc((size_t)H_D * IN_D * 2);
  f16* w1t   = (f16*)alloc((size_t)H_D * IN_D * 2);
  f16* w2h   = (f16*)alloc((size_t)H_D * H_D * 2);
  f16* w3h   = (f16*)alloc((size_t)H_D * H_D * 2);
  f16* wah   = (f16*)alloc((size_t)OUT_D * H_D * 2);
  f16* a1    = (f16*)alloc((size_t)B_D * H_D * 2);
  f16* m1    = (f16*)alloc((size_t)B_D * H_D * 2);
  f16* a2    = (f16*)alloc((size_t)B_D * H_D * 2);
  f16* m2    = (f16*)alloc((size_t)B_D * H_D * 2);
  f16* a3    = (f16*)alloc((size_t)B_D * H_D * 2);
  f16* m3    = (f16*)alloc((size_t)B_D * H_D * 2);
  float* fo  = (float*)alloc((size_t)B_D * OUT_D * 4);
  float* nrm = (float*)alloc((size_t)B_D * 4);
  float* kv  = (float*)alloc(16);

  convert_kernel<<<(B_D * IN_D + 255) / 256, 256, 0, stream>>>(
      x, W1, W2, W3, Wa, k, xh, w1h, w1t, w2h, w3h, wah, kv);

  // forward: tiles = (M/16)*(N/16), 8 waves (tiles) per block
  layer_kernel<<<(B_D / 16) * (H_D / 16) / 8, 256, 0, stream>>>(
      xh, w1h, b1, a1, m1, B_D, H_D, IN_D);
  layer_kernel<<<(B_D / 16) * (H_D / 16) / 8, 256, 0, stream>>>(
      a1, w2h, b2, a2, m2, B_D, H_D, H_D);
  layer_kernel<<<(B_D / 16) * (H_D / 16) / 8, 256, 0, stream>>>(
      a2, w3h, b3, a3, m3, B_D, H_D, H_D);
  head_kernel<<<(B_D / 16) * (OUT_D / 16) / 8, 256, 0, stream>>>(
      a3, wah, ba, fo, B_D, OUT_D, H_D);

  // per-sample Jacobian norms (the 137 GFLOP part)
  jac_kernel<<<B_D, 256, 0, stream>>>(w2h, w3h, w1t, m1, m2, m3, nrm);

  final_kernel<<<(B_D * OUT_D) / 256, 256, 0, stream>>>(fo, nrm, kv,
                                                        (float*)d_out);
}
